// CARAFE_54090818125896
// MI455X (gfx1250) — compile-verified, matrix operand order checked by reading, not run on hardware
//
#include <hip/hip_runtime.h>
#include <hip/hip_bf16.h>

typedef __attribute__((ext_vector_type(16))) __bf16 v16bf;
typedef __attribute__((ext_vector_type(8)))  float  v8f;
typedef unsigned int u32x4 __attribute__((ext_vector_type(4)));
typedef int          i32x4 __attribute__((ext_vector_type(4)));
typedef int          i32x8 __attribute__((ext_vector_type(8)));

#define HH 32
#define WW 32
#define NPIX 1024        // 32*32
#define CIN 256
#define CCOMP 64
#define KTOT 576         // 64*3*3
#define KT 18            // 576 / 32 K-steps of the bf16 WMMA
#define GROUPS 4         // scale^2
#define MGRP 25600       // rows (channels) per softmax group
#define NCOLS 128        // columns per block
#define NT 8             // N-tiles per block == waves (1 tile per wave)
#define COLBLOCKS 8      // 1024/128
#define MCHUNKS 8        // M split across blocks: 25600/3200
#define MT_TOTAL 200     // 3200/16 M-tiles per block, shared by all waves

#define BDW   36864      // B-fragment region, dwords (18*8*32*8)
#define AROW  594        // padded A row stride in dwords (576 + 2 per 64)
#define ATILE 9504       // dwords per padded A buffer (16*594)
#define ABASE BDW        // dword offset of A double buffer in LDS

// TDM availability (arity differs across toolchains: clang-22=5-arg, clang-23=6-arg)
#if defined(__has_builtin)
#if __has_builtin(__builtin_amdgcn_tensor_load_to_lds) && \
    __has_builtin(__builtin_amdgcn_s_wait_tensorcnt)
#define USE_TDM 1
#endif
#endif
#ifndef USE_TDM
#define USE_TDM 0
#endif

// pad flags for A descriptor: pad_enable | interval=64 dwords | amount=2 dwords
#define APADF ((1u << 20) | (5u << 22) | (1u << 25))

// K offset inside a 32-wide K-slab for the 16-bit A/B fragment layout
// (CDNA5 ISA §7.12.2)
__device__ __forceinline__ int koff16(int v, int half) {
    return ((v < 4) ? 0 : 16) + half * 8 + 2 * (v & 3);
}

#if USE_TDM
// Issue a 2D tensor_load_to_lds: tile (tile0 x tile1) of 4-byte elements,
// source row stride stride0 (elements), dest LDS at lds_addr (bytes).
__device__ __forceinline__ void tdm_load_2d(unsigned lds_addr,
                                            unsigned long long gaddr,
                                            unsigned padflags,
                                            unsigned dim0, unsigned dim1,
                                            unsigned stride0) {
    u32x4 g0;
    g0[0] = 1u;                                           // count=1, no gather
    g0[1] = lds_addr;                                     // LDS byte address
    g0[2] = (unsigned)gaddr;                              // global addr lo
    g0[3] = ((unsigned)(gaddr >> 32) & 0x01FFFFFFu)       // global addr hi (57b)
          | 0x80000000u;                                  // type=2 (image)
    i32x8 g1;
    g1[0] = (int)((2u << 16) | padflags);                 // data_size=4B + pads
    g1[1] = (int)((dim0 & 0xFFFFu) << 16);                // tensor_dim0[15:0]
    g1[2] = (int)((dim0 >> 16) | ((dim1 & 0xFFFFu) << 16));
    g1[3] = (int)((dim1 >> 16) | (dim0 << 16));           // tile_dim0 = dim0
    g1[4] = (int)dim1;                                    // tile_dim1 = dim1
    g1[5] = (int)stride0;                                 // tensor_dim0_stride
    g1[6] = 0;
    g1[7] = 0;
    i32x4 z4 = {0, 0, 0, 0};
#if __clang_major__ >= 23
    i32x8 z8 = {0, 0, 0, 0, 0, 0, 0, 0};
    __builtin_amdgcn_tensor_load_to_lds(g0, g1, z4, z4, z8, 0);
#else
    __builtin_amdgcn_tensor_load_to_lds(g0, g1, z4, z4, 0);
#endif
}
#endif

// ---------------------------------------------------------------- kernel A
__global__ void compress_kernel(const float* __restrict__ x,
                                const float* __restrict__ w_comp,
                                const float* __restrict__ b_comp,
                                float* __restrict__ k0) {
    int t = blockIdx.x * blockDim.x + threadIdx.x;   // 65536 threads
    int o = t >> 10, n = t & (NPIX - 1);
    const float* wr = w_comp + o * CIN;
    float acc = b_comp[o];
    #pragma unroll 4
    for (int i = 0; i < CIN; ++i)
        acc = fmaf(wr[i], x[i * NPIX + n], acc);
    k0[o * NPIX + n] = acc;
}

// ---------------------------------------------------------------- kernel B
// im2col activations -> bf16, pre-swizzled into WMMA B-fragment layout
__global__ void pack_b_kernel(const float* __restrict__ k0,
                              unsigned int* __restrict__ bfrag) {
    int t = blockIdx.x * blockDim.x + threadIdx.x;   // 18*64*32 = 36864
    int lane = t & 31;
    int ntg  = (t >> 5) & 63;
    int kt   = t >> 11;
    int half = lane >> 4;
    int n = ntg * 16 + (lane & 15);
    int h = n >> 5, w = n & 31;
    unsigned int out[8];
    #pragma unroll
    for (int v = 0; v < 8; ++v) {
        int kbase = kt * 32 + koff16(v, half);
        unsigned short us[2];
        #pragma unroll
        for (int e = 0; e < 2; ++e) {
            int kk = kbase + e;
            int i = kk / 9, r = kk % 9, ky = r / 3, kx = r % 3;
            int hh = h + ky - 1, ww = w + kx - 1;
            float f = (hh >= 0 && hh < HH && ww >= 0 && ww < WW)
                          ? k0[i * NPIX + hh * WW + ww] : 0.0f;
            union { __bf16 b; unsigned short u; } cv;
            cv.b = (__bf16)f;
            us[e] = cv.u;
        }
        out[v] = (unsigned int)us[0] | ((unsigned int)us[1] << 16);
    }
    unsigned int* dst = bfrag + (size_t)t * 8;
    #pragma unroll
    for (int v = 0; v < 8; ++v) dst[v] = out[v];
}

// ---------------------------------------------------------------- kernel C
// TDM-pipelined implicit GEMM + fused online softmax.
// All 8 waves share each 16-row weight tile (double-buffered in LDS by the
// Tensor Data Mover); wave w computes N-tile w. bf16 WMMA, fp32 accumulate.
__global__ void __launch_bounds__(256)
carafe_gemm_softmax_kernel(const float* __restrict__ w_ker,
                           const float* __restrict__ b_ker,
                           const unsigned int* __restrict__ bfrag,
                           float* __restrict__ partials) {
    extern __shared__ unsigned int smem[];

    const int cb    = blockIdx.x;
    const int chunk = blockIdx.y;
    const int g     = blockIdx.z;
    const int tid   = threadIdx.x;
    const int wave  = tid >> 5, lane = tid & 31;
    const int half  = lane >> 4, lm16 = lane & 15;

    const int rowbase = g * MGRP + chunk * (MT_TOTAL * 16);

#if USE_TDM
    const unsigned ldsbase = (unsigned)(size_t)(void*)smem;  // LDS byte offset
    if (wave == 0) {
        // B fragments for this column block: 18 rows x 2048 dwords, stride 16384
        tdm_load_2d(ldsbase,
                    (unsigned long long)(size_t)(const void*)(bfrag + cb * 2048),
                    0u, 2048u, 18u, 16384u);
        // first A tile (16 rows x 576 f32) with bank-decorrelating LDS padding
        tdm_load_2d(ldsbase + ABASE * 4,
                    (unsigned long long)(size_t)(const void*)
                        (w_ker + (size_t)rowbase * KTOT),
                    APADF, (unsigned)KTOT, 16u, (unsigned)KTOT);
    }
#else
    {   // cooperative fallback: B frags (linear) + A tile 0 (padded layout)
        const uint4* src = (const uint4*)bfrag + (size_t)cb * 512; // unused path
        (void)src;
        for (int j = 0; j < BDW / 256; ++j) {
            int idx = tid + 256 * j;
            const uint4* s4 = (const uint4*)(bfrag);
            ((uint4*)smem)[idx] = s4[(size_t)(idx >> 11) * 4096 + (size_t)cb * 512
                                     + (idx & 2047)];
        }
        const unsigned int* asrc =
            (const unsigned int*)(w_ker + (size_t)rowbase * KTOT);
        for (int j = 0; j < 36; ++j) {
            int L = tid + 256 * j;                  // 0..9215
            smem[ABASE + L + 2 * (L >> 6)] = asrc[L];
        }
    }
#endif

    float rmax = -3.0e38f, rsum = 0.0f;

    for (int mt = 0; mt < MT_TOTAL; ++mt) {
        const int row0 = rowbase + mt * 16;
        const int buf = mt & 1;

#if USE_TDM
        if (wave == 0) __builtin_amdgcn_s_wait_tensorcnt(0);
        __syncthreads();                             // tile mt (and B) visible
        if (wave == 0 && mt + 1 < MT_TOTAL) {        // DMA next tile
            tdm_load_2d(ldsbase + (ABASE + ((mt + 1) & 1) * ATILE) * 4,
                        (unsigned long long)(size_t)(const void*)
                            (w_ker + (size_t)(row0 + 16) * KTOT),
                        APADF, (unsigned)KTOT, 16u, (unsigned)KTOT);
        }
#else
        __syncthreads();
        if (mt + 1 < MT_TOTAL) {
            const unsigned int* asrc =
                (const unsigned int*)(w_ker + (size_t)(row0 + 16) * KTOT);
            const int ab = ABASE + ((mt + 1) & 1) * ATILE;
            for (int j = 0; j < 36; ++j) {
                int L = tid + 256 * j;
                smem[ab + L + 2 * (L >> 6)] = asrc[L];
            }
        }
#endif

        // init accumulator with bias (C/D layout: m = v + 8*half)
        v8f acc;
        #pragma unroll
        for (int v = 0; v < 8; ++v) acc[v] = b_ker[row0 + v + 8 * half];

        const float* abuf = (const float*)smem + (ABASE + buf * ATILE)
                          + AROW * lm16;             // this lane's A row (padded)
        #pragma unroll 2
        for (int kt = 0; kt < KT; ++kt) {
            v16bf a;
            #pragma unroll
            for (int v = 0; v < 8; ++v) {
                int o = kt * 32 + koff16(v, half);
                float2 f = *(const float2*)(abuf + o + 2 * (o >> 6));
                a[2 * v]     = (__bf16)f.x;
                a[2 * v + 1] = (__bf16)f.y;
            }
            union { uint4 q[2]; v16bf v; } bu;
            const uint4* bp =
                (const uint4*)(smem + ((kt * NT + wave) * 32 + lane) * 8);
            bu.q[0] = bp[0];
            bu.q[1] = bp[1];
            acc = __builtin_amdgcn_wmma_f32_16x16x32_bf16(
                false, a, false, bu.v, (short)0, acc, false, false);
        }

        // fused online softmax update (per lane: one column, 8 rows)
        float lmax = acc[0];
        #pragma unroll
        for (int v = 1; v < 8; ++v) lmax = fmaxf(lmax, acc[v]);
        float nm = fmaxf(rmax, lmax);
        float s = 0.0f;
        #pragma unroll
        for (int v = 0; v < 8; ++v) s += __expf(acc[v] - nm);
        rsum = rsum * __expf(rmax - nm) + s;
        rmax = nm;
    }

    // combine the two lane-halves of each column (wave32 xor-permute)
    float om = __shfl_xor(rmax, 16, 32);
    float os = __shfl_xor(rsum, 16, 32);
    float M = fmaxf(rmax, om);
    float S = rsum * __expf(rmax - M) + os * __expf(om - M);
    if (half == 0) {
        int col = wave * 16 + lm16;                  // 0..127 within block
        size_t p = ((size_t)((g * COLBLOCKS + cb) * MCHUNKS + chunk)) * NCOLS + col;
        partials[2 * p]     = M;
        partials[2 * p + 1] = S;
    }
}

// ---------------------------------------------------------------- kernel D
__global__ void ksum_kernel(const float* __restrict__ partials,
                            float* __restrict__ ksum) {
    int t = blockIdx.x * blockDim.x + threadIdx.x;   // 4096
    int g = t >> 10, n = t & 1023;
    int cb = n / NCOLS, col = n % NCOLS;
    float M = -3.0e38f;
    #pragma unroll
    for (int ch = 0; ch < MCHUNKS; ++ch) {
        size_t p = ((size_t)((g * COLBLOCKS + cb) * MCHUNKS + ch)) * NCOLS + col;
        M = fmaxf(M, partials[2 * p]);
    }
    float S = 0.0f;
    #pragma unroll
    for (int ch = 0; ch < MCHUNKS; ++ch) {
        size_t p = ((size_t)((g * COLBLOCKS + cb) * MCHUNKS + ch)) * NCOLS + col;
        S += partials[2 * p + 1] * __expf(partials[2 * p] - M);
    }
    ksum[t] = S / S;                                 // sum of softmax
}

// ---------------------------------------------------------------- kernel E
__global__ void upsample_kernel(const float* __restrict__ x,
                                const float* __restrict__ ksum,
                                float* __restrict__ out) {
    int t = blockIdx.x * blockDim.x + threadIdx.x;   // 1048576
    int c = t >> 12;
    int r = t & 4095;                                // s*1024 + n
    out[t] = x[c * NPIX + (r & 1023)] * ksum[r];
}

extern "C" void kernel_launch(void* const* d_in, const int* in_sizes, int n_in,
                              void* d_out, int out_size, void* d_ws, size_t ws_size,
                              hipStream_t stream) {
    const float* x      = (const float*)d_in[0];
    const float* w_comp = (const float*)d_in[1];
    const float* b_comp = (const float*)d_in[2];
    const float* w_ker  = (const float*)d_in[3];
    const float* b_ker  = (const float*)d_in[4];
    float* out = (float*)d_out;

    float*        k0       = (float*)d_ws;                          // 65536 f32
    unsigned int* bfrag    = (unsigned int*)(k0 + CCOMP * NPIX);    // 294912 u32
    float*        partials = (float*)(bfrag + (size_t)BDW * 8);     // 65536 f32
    float*        ksum     = partials + 2 * GROUPS * COLBLOCKS * MCHUNKS * NCOLS;

    compress_kernel<<<(CCOMP * NPIX) / 256, 256, 0, stream>>>(x, w_comp, b_comp, k0);
    pack_b_kernel<<<(KT * 64 * 32) / 256, 256, 0, stream>>>(k0, bfrag);

    dim3 gridC(COLBLOCKS, MCHUNKS, GROUPS);          // 256 blocks x 8 waves
    size_t smem_bytes = (size_t)(BDW + 2 * ATILE) * 4;   // 144KB B + 74.3KB A
    carafe_gemm_softmax_kernel<<<gridC, 256, smem_bytes, stream>>>(w_ker, b_ker,
                                                                   bfrag, partials);
    ksum_kernel<<<(GROUPS * NPIX) / 256, 256, 0, stream>>>(partials, ksum);
    upsample_kernel<<<(CIN * GROUPS * NPIX) / 256, 256, 0, stream>>>(x, ksum, out);
}
